// TemplateAwareAttention_2826088481249
// MI455X (gfx1250) — matrix operand order repaired and checked by reading, not compiled, not run on hardware
//
#include <hip/hip_runtime.h>
#include <hip/hip_bf16.h>
#include <cstdint>
#include <cstddef>

// Problem constants (match reference)
#define B_  4
#define S_  2048
#define D_  1024
#define H_  16
#define HD_ 64
#define EPS_ 1e-5f

typedef __attribute__((ext_vector_type(16))) _Float16 v16h;
typedef __attribute__((ext_vector_type(8)))  _Float16 v8h;
typedef __attribute__((ext_vector_type(8)))  float    v8f;

__device__ __forceinline__ v16h frag_pair(const _Float16* p0, const _Float16* p1) {
    union { v16h v; v8h h[2]; } u;
    u.h[0] = *(const v8h*)p0;
    u.h[1] = *(const v8h*)p1;
    return u.v;
}

__device__ __forceinline__ v8f wmma_f16(v16h a, v16h b, v8f c) {
    return __builtin_amdgcn_wmma_f32_16x16x32_f16(false, a, false, b, (short)0, c, false, false);
}

// ---------------------------------------------------------------------------
// f32 -> f16 elementwise convert
// ---------------------------------------------------------------------------
__global__ void f32_to_f16_kernel(const float* __restrict__ src,
                                  _Float16* __restrict__ dst, int n) {
    int i = blockIdx.x * blockDim.x + threadIdx.x;
    if (i < n) dst[i] = (_Float16)src[i];
}

// ---------------------------------------------------------------------------
// C[M,N] = A[M,K](f16) @ W[N,K](f16)^T + bias[N]
// Block tile 128x128, 8 waves (2 along M x 4 along N), each wave 64x32,
// K-step 32 -> one v_wmma_f32_16x16x32_f16 per 16x16 subtile per step.
//
// OUT_MODE: 0 = f32 row-major [M][N]
//           1 = f16 row-major [M][N]
//           2 = f16 per-head transposed [B][H][HD][S]  (for V: reused 32x
//               by attention, so pay the scatter once here and make the
//               PV B-operand a contiguous ds_load_b128 pair)
// ---------------------------------------------------------------------------
#define GEMM_LDA 40   // 32 halves + 8 pad (80B rows, 16B aligned)

template <int OUT_MODE>
__global__ __launch_bounds__(256) void gemm_bias_wmma(
    const _Float16* __restrict__ A, const _Float16* __restrict__ W,
    const float* __restrict__ bias, void* __restrict__ Cout,
    int M, int N, int K)
{
    __shared__ __align__(16) _Float16 sA[128 * GEMM_LDA];
    __shared__ __align__(16) _Float16 sB[128 * GEMM_LDA];

    const int tid  = threadIdx.x;
    const int lane = tid & 31;
    const int wave = tid >> 5;
    const int wm   = wave & 1;       // 0..1  (64 rows each)
    const int wn   = wave >> 1;      // 0..3  (32 cols each)
    const int hi   = lane >> 4;      // lane half-group
    const int lr   = lane & 15;
    const int bm   = blockIdx.x * 128;
    const int bn   = blockIdx.y * 128;

    v8f acc[4][2];
    #pragma unroll
    for (int i = 0; i < 4; ++i)
        #pragma unroll
        for (int j = 0; j < 2; ++j)
            #pragma unroll
            for (int v = 0; v < 8; ++v) acc[i][j][v] = 0.f;

    for (int k0 = 0; k0 < K; k0 += 32) {
        __syncthreads();
        // Cooperative tile load: 128 rows x 32 halves, 4 threads/row, 2 passes
        {
            int row = tid >> 2;
            const int seg = (tid & 3) * 8;
            #pragma unroll
            for (int p = 0; p < 2; ++p, row += 64) {
                *(uint4*)&sA[row * GEMM_LDA + seg] =
                    *(const uint4*)&A[(size_t)(bm + row) * K + k0 + seg];
                *(uint4*)&sB[row * GEMM_LDA + seg] =
                    *(const uint4*)&W[(size_t)(bn + row) * K + k0 + seg];
                if (k0 + 32 < K) {
                    __builtin_prefetch(&A[(size_t)(bm + row) * K + k0 + 32 + seg], 0, 1);
                    __builtin_prefetch(&W[(size_t)(bn + row) * K + k0 + 32 + seg], 0, 1);
                }
            }
        }
        __syncthreads();

        // B fragments (K x N layout: lane = N col, contiguous K halves)
        v16h bf[2];
        #pragma unroll
        for (int ns = 0; ns < 2; ++ns) {
            const _Float16* p = &sB[(wn * 32 + ns * 16 + lr) * GEMM_LDA + hi * 16];
            bf[ns] = frag_pair(p, p + 8);
        }
        #pragma unroll
        for (int ms = 0; ms < 4; ++ms) {
            const _Float16* pa = &sA[(wm * 64 + ms * 16 + lr) * GEMM_LDA + hi * 16];
            v16h af = frag_pair(pa, pa + 8);   // K = hi*16..+15
            #pragma unroll
            for (int ns = 0; ns < 2; ++ns)
                acc[ms][ns] = wmma_f16(af, bf[ns], acc[ms][ns]);
        }
    }

    // Epilogue: C/D layout -> m = vgpr + 8*hi, n = lane&15
    #pragma unroll
    for (int ms = 0; ms < 4; ++ms) {
        #pragma unroll
        for (int ns = 0; ns < 2; ++ns) {
            const int n  = bn + wn * 32 + ns * 16 + lr;
            const float bv = bias[n];
            #pragma unroll
            for (int v = 0; v < 8; ++v) {
                const int m = bm + wm * 64 + ms * 16 + v + hi * 8;
                const float val = acc[ms][ns][v] + bv;
                if (OUT_MODE == 0) {
                    ((float*)Cout)[(size_t)m * N + n] = val;
                } else if (OUT_MODE == 1) {
                    ((_Float16*)Cout)[(size_t)m * N + n] = (_Float16)val;
                } else {
                    // [B][H][HD][S]: b = m/S, s = m%S, h = n/HD, hd = n%HD
                    const int b  = m >> 11;          // /S_ (2048)
                    const int s  = m & (S_ - 1);
                    const int h  = n >> 6;           // /HD_ (64)
                    const int hd = n & (HD_ - 1);
                    ((_Float16*)Cout)[(((size_t)(b * H_ + h) * HD_ + hd) * S_) + s] =
                        (_Float16)val;
                }
            }
        }
    }
}

// ---------------------------------------------------------------------------
// Flash attention, one (b,h,64-query-tile) per block, 4 waves (128 threads).
// scores = Q K^T / 8 (masked), online softmax, O = P V, f16 output.
// V comes in pre-transposed [B][H][HD][S] so both WMMA B-operands are
// contiguous 16B LDS reads (no scalar gathers).
// ---------------------------------------------------------------------------
#define KLD 72   // 64 halves + 8 pad
#define SLD 68   // 64 floats + 4 pad

__global__ __launch_bounds__(128) void flash_attn_wmma(
    const _Float16* __restrict__ Q, const _Float16* __restrict__ Kmat,
    const _Float16* __restrict__ Vt, const int* __restrict__ mask,
    _Float16* __restrict__ O)
{
    __shared__ __align__(16) _Float16 sK[64 * KLD];   // [key][hd]
    __shared__ __align__(16) _Float16 sV[64 * KLD];   // [hd][key]  (transposed)
    __shared__ __align__(16) _Float16 sP[64 * KLD];   // [query][key]
    __shared__ float sS[64 * SLD];
    __shared__ float stM[64], stL[64], stAlpha[64];
    __shared__ float red[128];

    const int tid  = threadIdx.x;
    const int lane = tid & 31;
    const int w    = tid >> 5;          // wave 0..3 -> query strip
    const int hi   = lane >> 4;
    const int lr   = lane & 15;

    const int nQt = S_ / 64;
    const int bh  = blockIdx.x / nQt;
    const int qt  = blockIdx.x % nQt;
    const int b   = bh / H_;
    const int h   = bh % H_;
    const size_t rowBase = (size_t)b * S_;
    const int hoff = h * HD_;
    const _Float16* VtBase = Vt + (size_t)bh * HD_ * S_;

    for (int r = tid; r < 64; r += 128) { stM[r] = -1e30f; stL[r] = 0.f; }

    // Preload Q fragments (A operand), queries = qt*64 + w*16 + lr
    v16h aQ[2];
    {
        const _Float16* qp = &Q[(rowBase + qt * 64 + w * 16 + lr) * D_ + hoff];
        #pragma unroll
        for (int ks = 0; ks < 2; ++ks)
            aQ[ks] = frag_pair(qp + ks * 32 + hi * 16, qp + ks * 32 + 8 + hi * 16);
    }

    v8f oAcc[4];
    #pragma unroll
    for (int ns = 0; ns < 4; ++ns)
        #pragma unroll
        for (int v = 0; v < 8; ++v) oAcc[ns][v] = 0.f;

    for (int kt = 0; kt < S_ / 64; ++kt) {
        __syncthreads();
        // Stage K [key][hd] and V^T [hd][key] tiles; both coalesced uint4
        for (int c = tid; c < 64 * 8; c += 128) {
            const int row = c >> 3, seg = (c & 7) * 8;
            *(uint4*)&sK[row * KLD + seg] =
                *(const uint4*)&Kmat[(rowBase + kt * 64 + row) * D_ + hoff + seg];
            *(uint4*)&sV[row * KLD + seg] =
                *(const uint4*)&VtBase[(size_t)row * S_ + kt * 64 + seg];
        }
        __syncthreads();

        // ---- scores: 4 key subtiles x 2 K-steps of WMMA ----
        #pragma unroll
        for (int ns = 0; ns < 4; ++ns) {
            v8f s;
            #pragma unroll
            for (int v = 0; v < 8; ++v) s[v] = 0.f;
            #pragma unroll
            for (int ks = 0; ks < 2; ++ks) {
                const _Float16* bp = &sK[(ns * 16 + lr) * KLD + ks * 32 + hi * 16];
                s = wmma_f16(aQ[ks], frag_pair(bp, bp + 8), s);
            }
            const int key = kt * 64 + ns * 16 + lr;
            const int mv  = mask[b * S_ + key];
            #pragma unroll
            for (int v = 0; v < 8; ++v) {
                float sv = s[v] * 0.125f;           // 1/sqrt(64)
                if (mv == 0) sv = -10000.f;
                sS[(w * 16 + v + hi * 8) * SLD + ns * 16 + lr] = sv;
            }
        }
        __syncthreads();

        // ---- online softmax (2 threads per row) ----
        {
            const int r = tid >> 1, hc = tid & 1;
            const float* srow = &sS[r * SLD + hc * 32];
            float mx = -1e30f;
            #pragma unroll
            for (int c = 0; c < 32; ++c) mx = fmaxf(mx, srow[c]);
            red[tid] = mx;
            __syncthreads();
            mx = fmaxf(red[r * 2], red[r * 2 + 1]);
            const float mOld = stM[r];
            const float mNew = fmaxf(mOld, mx);
            float sum = 0.f;
            _Float16* prow = &sP[r * KLD + hc * 32];
            #pragma unroll
            for (int c = 0; c < 32; ++c) {
                const float p = __expf(srow[c] - mNew);
                prow[c] = (_Float16)p;
                sum += p;
            }
            __syncthreads();
            red[tid] = sum;
            __syncthreads();
            if (hc == 0) {
                const float alpha = __expf(mOld - mNew);
                stAlpha[r] = alpha;
                stM[r]     = mNew;
                stL[r]     = stL[r] * alpha + red[r * 2] + red[r * 2 + 1];
            }
        }
        __syncthreads();

        // ---- rescale accumulator, then O += P V via WMMA ----
        #pragma unroll
        for (int ns = 0; ns < 4; ++ns)
            #pragma unroll
            for (int v = 0; v < 8; ++v)
                oAcc[ns][v] *= stAlpha[w * 16 + v + hi * 8];

        const _Float16* pp = &sP[(w * 16 + lr) * KLD];
        #pragma unroll
        for (int ks = 0; ks < 2; ++ks) {
            v16h af = frag_pair(pp + ks * 32 + hi * 16, pp + ks * 32 + 8 + hi * 16);
            #pragma unroll
            for (int ns = 0; ns < 4; ++ns) {
                // B operand from transposed V: lane = hd col, contiguous keys
                const _Float16* bp = &sV[(ns * 16 + lr) * KLD + ks * 32 + hi * 16];
                oAcc[ns] = wmma_f16(af, frag_pair(bp, bp + 8), oAcc[ns]);
            }
        }
    }

    __syncthreads();
    // Normalize by running sum and store f16 for the O-projection GEMM
    #pragma unroll
    for (int ns = 0; ns < 4; ++ns)
        #pragma unroll
        for (int v = 0; v < 8; ++v) {
            const int m = w * 16 + v + hi * 8;
            const float val = oAcc[ns][v] / stL[m];
            O[(rowBase + qt * 64 + m) * D_ + hoff + ns * 16 + lr] = (_Float16)val;
        }
}

// ---------------------------------------------------------------------------
// out = LayerNorm(x + proj) * gamma + beta ; one row per block (256 threads)
// ---------------------------------------------------------------------------
__global__ __launch_bounds__(256) void add_layernorm_kernel(
    const float* __restrict__ x, const float* __restrict__ proj,
    const float* __restrict__ gamma, const float* __restrict__ beta,
    float* __restrict__ out)
{
    __shared__ float red[256];
    const int tid = threadIdx.x;
    const size_t row = blockIdx.x;
    const float* xp = x + row * D_;
    const float* pp = proj + row * D_;

    float y[4];
    float s = 0.f;
    #pragma unroll
    for (int i = 0; i < 4; ++i) {
        y[i] = xp[tid + i * 256] + pp[tid + i * 256];
        s += y[i];
    }
    red[tid] = s; __syncthreads();
    for (int o = 128; o > 0; o >>= 1) {
        if (tid < o) red[tid] += red[tid + o];
        __syncthreads();
    }
    const float mu = red[0] * (1.f / D_);
    __syncthreads();

    float vs = 0.f;
    #pragma unroll
    for (int i = 0; i < 4; ++i) { const float d = y[i] - mu; vs += d * d; }
    red[tid] = vs; __syncthreads();
    for (int o = 128; o > 0; o >>= 1) {
        if (tid < o) red[tid] += red[tid + o];
        __syncthreads();
    }
    const float inv = rsqrtf(red[0] * (1.f / D_) + EPS_);

    #pragma unroll
    for (int i = 0; i < 4; ++i) {
        const int c = tid + i * 256;
        out[row * D_ + c] = (y[i] - mu) * inv * gamma[c] + beta[c];
    }
}

// ---------------------------------------------------------------------------
// Host-side orchestration
// ---------------------------------------------------------------------------
extern "C" void kernel_launch(void* const* d_in, const int* in_sizes, int n_in,
                              void* d_out, int out_size, void* d_ws, size_t ws_size,
                              hipStream_t stream) {
    (void)in_sizes; (void)n_in; (void)out_size; (void)ws_size;

    const float* x    = (const float*)d_in[0];
    const int*   mask = (const int*)d_in[2];
    const float* Wq   = (const float*)d_in[3];
    const float* bq   = (const float*)d_in[4];
    const float* Wk   = (const float*)d_in[5];
    const float* bk   = (const float*)d_in[6];
    const float* Wv   = (const float*)d_in[7];
    const float* bv   = (const float*)d_in[8];
    const float* Wo   = (const float*)d_in[9];
    const float* bo   = (const float*)d_in[10];
    const float* gamma= (const float*)d_in[11];
    const float* beta = (const float*)d_in[12];
    float* out = (float*)d_out;

    const size_t ND = (size_t)B_ * S_ * D_;   // 8,388,608
    const size_t DD = (size_t)D_ * D_;        // 1,048,576
    const int    M  = B_ * S_;                // 8192
    const int    N  = D_;
    const int    K  = D_;

    char* ws = (char*)d_ws;
    auto carve = [&](size_t bytes) -> char* {
        char* p = ws;
        ws += (bytes + 255) & ~(size_t)255;
        return p;
    };
    _Float16* xh    = (_Float16*)carve(ND * 2);
    _Float16* Wqh   = (_Float16*)carve(DD * 2);
    _Float16* Wkh   = (_Float16*)carve(DD * 2);
    _Float16* Wvh   = (_Float16*)carve(DD * 2);
    _Float16* Woh   = (_Float16*)carve(DD * 2);
    _Float16* Qh    = (_Float16*)carve(ND * 2);
    _Float16* Kh    = (_Float16*)carve(ND * 2);
    _Float16* Vth   = (_Float16*)carve(ND * 2);   // V, transposed per head
    _Float16* Ah    = (_Float16*)carve(ND * 2);   // attention output (f16)
    float*    proj  = (float*)carve(ND * 4);      // O-projection (f32)

    // 1) Convert activations & weights to f16 for the WMMA tensor path
    f32_to_f16_kernel<<<(int)(ND / 256), 256, 0, stream>>>(x,  xh,  (int)ND);
    f32_to_f16_kernel<<<(int)(DD / 256), 256, 0, stream>>>(Wq, Wqh, (int)DD);
    f32_to_f16_kernel<<<(int)(DD / 256), 256, 0, stream>>>(Wk, Wkh, (int)DD);
    f32_to_f16_kernel<<<(int)(DD / 256), 256, 0, stream>>>(Wv, Wvh, (int)DD);
    f32_to_f16_kernel<<<(int)(DD / 256), 256, 0, stream>>>(Wo, Woh, (int)DD);

    // 2) QKV projections (Q,K row-major f16; V transposed-per-head f16)
    dim3 ggrid(M / 128, N / 128);
    gemm_bias_wmma<1><<<ggrid, 256, 0, stream>>>(xh, Wqh, bq, Qh,  M, N, K);
    gemm_bias_wmma<1><<<ggrid, 256, 0, stream>>>(xh, Wkh, bk, Kh,  M, N, K);
    gemm_bias_wmma<2><<<ggrid, 256, 0, stream>>>(xh, Wvh, bv, Vth, M, N, K);

    // 3) Flash attention: B*H*(S/64) blocks, 128 threads (4 waves)
    flash_attn_wmma<<<B_ * H_ * (S_ / 64), 128, 0, stream>>>(Qh, Kh, Vth, mask, Ah);

    // 4) Output projection (f32 output)
    gemm_bias_wmma<0><<<ggrid, 256, 0, stream>>>(Ah, Woh, bo, proj, M, N, K);

    // 5) Residual + LayerNorm
    add_layernorm_kernel<<<M, 256, 0, stream>>>(x, proj, gamma, beta, out);
}
